// NN_RELU_Conditioner_T_79697413144877
// MI455X (gfx1250) — compile-verified, matrix operand order checked by reading, not run on hardware
//
#include <hip/hip_runtime.h>
#include <hip/hip_bf16.h>
#include <stdint.h>

// ---------------------------------------------------------------------------
// Problem constants (match reference)
// ---------------------------------------------------------------------------
constexpr int C_   = 4;
constexpr int IN_  = 256;
constexpr int W_   = 512;
constexpr int OUT_ = 128;
constexpr int N_   = 32768;
constexpr int TILE = 32;              // samples per block (2 M-tiles)

// ---------------------------------------------------------------------------
// Workspace layout (bytes)
// ---------------------------------------------------------------------------
constexpr size_t OFF_WINT   = 0;                                   // [C][W][IN]  bf16
constexpr size_t SZ_WINT    = (size_t)C_ * W_ * IN_ * 2;           // 1 MB
constexpr size_t OFF_WHT    = OFF_WINT + SZ_WINT;                  // [C][2][W][W] bf16
constexpr size_t SZ_WHT     = (size_t)C_ * 2 * W_ * W_ * 2;        // 4 MB
constexpr size_t OFF_WOUTT  = OFF_WHT + SZ_WHT;                    // [C][OUT][W] bf16
constexpr size_t SZ_WOUTT   = (size_t)C_ * OUT_ * W_ * 2;          // 0.5 MB
constexpr size_t OFF_COUNTS = OFF_WOUTT + SZ_WOUTT;                // C ints (padded)
constexpr size_t OFF_IDX    = OFF_COUNTS + 256;                    // [C][N] ints

// ---------------------------------------------------------------------------
// Types for WMMA
// ---------------------------------------------------------------------------
typedef __attribute__((ext_vector_type(16))) __bf16 v16bf;
typedef __attribute__((ext_vector_type(8)))  float  v8f;

struct Frag32 { uint4 a; uint4 b; };   // 32 bytes == v16bf

__device__ __forceinline__ unsigned short f32_to_bf16(float f) {
    uint32_t u = __builtin_bit_cast(uint32_t, f);
    uint32_t r = u + 0x7FFFu + ((u >> 16) & 1u);   // round-to-nearest-even
    return (unsigned short)(r >> 16);
}

// Load one 16x32 bf16 fragment (A-style layout; Bᵀ uses identical layout).
// Lanes 0-15:  row = row0+lane,    K = [k0..k0+7]  and [k0+16..k0+23]
// Lanes 16-31: row = row0+lane-16, K = [k0+8..k0+15] and [k0+24..k0+31]
__device__ __forceinline__ v16bf load_frag(const unsigned short* base, int stride,
                                           int row0, int k0, int lane) {
    int half = lane >> 4;
    int rr   = lane & 15;
    const unsigned short* p = base + (size_t)(row0 + rr) * stride + k0 + half * 8;
    Frag32 f;
    f.a = *(const uint4*)(p);
    f.b = *(const uint4*)(p + 16);
    return __builtin_bit_cast(v16bf, f);
}

// ---------------------------------------------------------------------------
// Kernel 0: zero per-class counters
// ---------------------------------------------------------------------------
__global__ void zero_counts_kernel(int* counts) {
    if (threadIdx.x < C_) counts[threadIdx.x] = 0;
}

// ---------------------------------------------------------------------------
// Kernel 1: f32 [B][K][N] -> bf16 transposed [B][N][K]
// ---------------------------------------------------------------------------
__global__ void transpose_bf16_kernel(const float* __restrict__ in,
                                      unsigned short* __restrict__ out,
                                      int B, int K, int Nn) {
    size_t total = (size_t)B * K * Nn;
    size_t gid = (size_t)blockIdx.x * blockDim.x + threadIdx.x;
    size_t gstride = (size_t)gridDim.x * blockDim.x;
    for (; gid < total; gid += gstride) {
        size_t b   = gid / ((size_t)Nn * K);
        size_t rem = gid % ((size_t)Nn * K);
        size_t n   = rem / K;
        size_t k   = rem % K;
        out[gid] = f32_to_bf16(in[(b * K + k) * Nn + n]);
    }
}

// ---------------------------------------------------------------------------
// Kernel 2: partition samples by class
// ---------------------------------------------------------------------------
__global__ void partition_kernel(const float* __restrict__ x,
                                 int* counts, int* idxlist) {
    int i = blockIdx.x * blockDim.x + threadIdx.x;
    if (i >= N_) return;
    int c = (int)x[(size_t)i * (IN_ + 1)];
    c = c < 0 ? 0 : (c >= C_ ? C_ - 1 : c);
    int pos = atomicAdd(counts + c, 1);
    idxlist[(size_t)c * N_ + pos] = i;
}

// ---------------------------------------------------------------------------
// GEMM layer: oLds[TILE][NOUT] = act(aLds[TILE][K] * wT[NOUT][K]^T + bias)
// 8 waves, each owning BOTH M-tiles of one column strip. B fragments are
// software-pipelined (prefetch distance 1) so a load is always in flight
// while the two WMMAs of the previous K-step execute.
// ---------------------------------------------------------------------------
template <int K, int NOUT, bool RELU>
__device__ __forceinline__ void gemm_layer(const unsigned short* aLds,
                                           const unsigned short* __restrict__ wT,
                                           const float* __restrict__ bias,
                                           unsigned short* oLds,
                                           int lane, int wave) {
    constexpr int TPW = NOUT / 16 / 8;          // N-tiles per wave (4 or 1)
    constexpr int KT  = K / 32;
    const int half = lane >> 4;
    const int nc   = lane & 15;
    for (int nt = wave * TPW; nt < (wave + 1) * TPW; ++nt) {
        const int n0 = nt * 16;
        v8f acc0 = {};
        v8f acc1 = {};
        v16bf bcur = load_frag(wT, K, n0, 0, lane);
        #pragma unroll
        for (int kt = 0; kt < KT; ++kt) {
            v16bf bnext = bcur;
            if (kt + 1 < KT) bnext = load_frag(wT, K, n0, (kt + 1) * 32, lane);
            v16bf a0 = load_frag(aLds, W_, 0,  kt * 32, lane);
            v16bf a1 = load_frag(aLds, W_, 16, kt * 32, lane);
            acc0 = __builtin_amdgcn_wmma_f32_16x16x32_bf16(
                false, a0, false, bcur, (short)0, acc0, false, false);
            acc1 = __builtin_amdgcn_wmma_f32_16x16x32_bf16(
                false, a1, false, bcur, (short)0, acc1, false, false);
            bcur = bnext;
        }
        const int col = n0 + nc;
        const float bv = bias[col];
        #pragma unroll
        for (int j = 0; j < 8; ++j) {
            int row = half * 8 + j;
            float v0 = acc0[j] + bv;
            float v1 = acc1[j] + bv;
            if (RELU) { v0 = fmaxf(v0, 0.0f); v1 = fmaxf(v1, 0.0f); }
            oLds[row * W_ + col]        = f32_to_bf16(v0);
            oLds[(row + 16) * W_ + col] = f32_to_bf16(v1);
        }
    }
}

// ---------------------------------------------------------------------------
// Kernel 3: per-class MLP over a tile of gathered samples
// ---------------------------------------------------------------------------
__global__ __launch_bounds__(256) void mlp_kernel(
        const float* __restrict__ x,
        const float* __restrict__ b_in,
        const float* __restrict__ b_h,
        const float* __restrict__ b_out,
        const unsigned short* __restrict__ WinT,
        const unsigned short* __restrict__ WhT,
        const unsigned short* __restrict__ WoutT,
        const int* __restrict__ counts,
        const int* __restrict__ idxlist,
        float* __restrict__ out) {
    __shared__ unsigned short bufA[TILE * W_];   // 32 KB
    __shared__ unsigned short bufB[TILE * W_];   // 32 KB

    const int c   = blockIdx.y;
    const int cnt = counts[c];
    const int n0  = blockIdx.x * TILE;
    if (n0 >= cnt) return;
    const int rows = min(TILE, cnt - n0);
    const int* list = idxlist + (size_t)c * N_ + n0;

    const int tid  = threadIdx.x;
    const int lane = tid & 31;
    const int wave = tid >> 5;

    // ---- gather features (f32 -> bf16) into bufA[r][0..IN_) ----
    {
        int r     = tid >> 3;          // 8 threads per row
        int kbase = (tid & 7) * 32;    // 32 contiguous cols each
        if (r < rows) {
            int idx = list[r];
            const float* src = x + (size_t)idx * (IN_ + 1) + 1 + kbase;
            #pragma unroll 8
            for (int k = 0; k < 32; ++k)
                bufA[r * W_ + kbase + k] = f32_to_bf16(src[k]);
        } else {
            #pragma unroll 8
            for (int k = 0; k < 32; ++k)
                bufA[r * W_ + kbase + k] = 0;
        }
    }
    __syncthreads();

    // ---- layer 0: 256 -> 512, relu ----
    gemm_layer<IN_, W_, true>(bufA, WinT + (size_t)c * W_ * IN_,
                              b_in + (size_t)c * W_, bufB, lane, wave);
    __syncthreads();
    // ---- layer 1: 512 -> 512, relu ----
    gemm_layer<W_, W_, true>(bufB, WhT + (size_t)(c * 2 + 0) * W_ * W_,
                             b_h + (size_t)(c * 2 + 0) * W_, bufA, lane, wave);
    __syncthreads();
    // ---- layer 2: 512 -> 512, relu ----
    gemm_layer<W_, W_, true>(bufA, WhT + (size_t)(c * 2 + 1) * W_ * W_,
                             b_h + (size_t)(c * 2 + 1) * W_, bufB, lane, wave);
    __syncthreads();

    // ---- output layer: 512 -> 128, scatter f32 rows by original index ----
    {
        const int half = lane >> 4;
        const int nc   = lane & 15;
        const float* bo = b_out + (size_t)c * OUT_;
        const unsigned short* wT = WoutT + (size_t)c * OUT_ * W_;
        const int ncol0 = wave * 16;             // 8 waves x 1 N-tile
        constexpr int KT = W_ / 32;
        v8f acc0 = {};
        v8f acc1 = {};
        v16bf bcur = load_frag(wT, W_, ncol0, 0, lane);
        #pragma unroll
        for (int kt = 0; kt < KT; ++kt) {
            v16bf bnext = bcur;
            if (kt + 1 < KT) bnext = load_frag(wT, W_, ncol0, (kt + 1) * 32, lane);
            v16bf a0 = load_frag(bufB, W_, 0,  kt * 32, lane);
            v16bf a1 = load_frag(bufB, W_, 16, kt * 32, lane);
            acc0 = __builtin_amdgcn_wmma_f32_16x16x32_bf16(
                false, a0, false, bcur, (short)0, acc0, false, false);
            acc1 = __builtin_amdgcn_wmma_f32_16x16x32_bf16(
                false, a1, false, bcur, (short)0, acc1, false, false);
            bcur = bnext;
        }
        const int col = ncol0 + nc;
        const float bv = bo[col];
        #pragma unroll
        for (int j = 0; j < 8; ++j) {
            int r0 = half * 8 + j;
            int r1 = r0 + 16;
            if (r0 < rows) {
                int idx = list[r0];
                out[(size_t)idx * OUT_ + col] = acc0[j] + bv;
            }
            if (r1 < rows) {
                int idx = list[r1];
                out[(size_t)idx * OUT_ + col] = acc1[j] + bv;
            }
        }
    }
}

// ---------------------------------------------------------------------------
// Host launcher
// ---------------------------------------------------------------------------
extern "C" void kernel_launch(void* const* d_in, const int* in_sizes, int n_in,
                              void* d_out, int out_size, void* d_ws, size_t ws_size,
                              hipStream_t stream) {
    const float* x     = (const float*)d_in[0];
    const float* W_in  = (const float*)d_in[1];
    const float* b_in  = (const float*)d_in[2];
    const float* W_h   = (const float*)d_in[3];
    const float* b_h   = (const float*)d_in[4];
    const float* W_out = (const float*)d_in[5];
    const float* b_out = (const float*)d_in[6];
    float* out = (float*)d_out;

    uint8_t* ws = (uint8_t*)d_ws;
    unsigned short* WinT  = (unsigned short*)(ws + OFF_WINT);
    unsigned short* WhT   = (unsigned short*)(ws + OFF_WHT);
    unsigned short* WoutT = (unsigned short*)(ws + OFF_WOUTT);
    int* counts  = (int*)(ws + OFF_COUNTS);
    int* idxlist = (int*)(ws + OFF_IDX);

    // 0) reset counters
    zero_counts_kernel<<<1, 64, 0, stream>>>(counts);

    // 1) weight conversion + transpose to bf16
    {
        size_t t1 = (size_t)C_ * IN_ * W_;
        transpose_bf16_kernel<<<(int)((t1 + 255) / 256), 256, 0, stream>>>(
            W_in, WinT, C_, IN_, W_);
        size_t t2 = (size_t)C_ * 2 * W_ * W_;
        transpose_bf16_kernel<<<(int)((t2 + 255) / 256), 256, 0, stream>>>(
            W_h, WhT, C_ * 2, W_, W_);
        size_t t3 = (size_t)C_ * W_ * OUT_;
        transpose_bf16_kernel<<<(int)((t3 + 255) / 256), 256, 0, stream>>>(
            W_out, WoutT, C_, W_, OUT_);
    }

    // 2) partition samples by class
    partition_kernel<<<N_ / 256, 256, 0, stream>>>(x, counts, idxlist);

    // 3) routed MLP
    dim3 grid(N_ / TILE, C_);
    mlp_kernel<<<grid, 256, 0, stream>>>(x, b_in, b_h, b_out,
                                         WinT, WhT, WoutT,
                                         counts, idxlist, out);
}
